// SingleHeadedAttention_85220741087270
// MI455X (gfx1250) — matrix operand order repaired and checked by reading, not compiled, not run on hardware
//
#include <hip/hip_runtime.h>
#include <math.h>

// ---------------------------------------------------------------------------
// MI455X (gfx1250) fused single-headed laplacian attention.
//   B=8, S=2048, D=1024, DQK=128, DV=1024. All matmuls in bf16 WMMA
//   (V_WMMA_F32_16X16X32_BF16, f32 accumulate); activations in f32 VALU.
//   Flash-style: the S x S attn matrix lives only as 16x128 bf16 LDS tiles.
//   Software-pipelined: fragment loads for step k+1 are issued before the
//   WMMA chain of step k, so s_wait_loadcnt waits cover already-landed loads.
// ---------------------------------------------------------------------------

#define DIM   1024
#define DQK   128
#define DV    1024
#define BATCH 8
#define SEQ   2048
#define ROWS  (BATCH * SEQ)   // 16384

typedef __attribute__((ext_vector_type(16))) __bf16 v16bf;
typedef __attribute__((ext_vector_type(8)))  float  v8f;

union FragB { v16bf v; unsigned int u[8]; };

__device__ __forceinline__ v8f v8f_zero() {
  v8f z;
#pragma unroll
  for (int i = 0; i < 8; ++i) z[i] = 0.0f;
  return z;
}

__device__ __forceinline__ unsigned short f2bf(float f) {
  unsigned int u = __float_as_uint(f);
  u += 0x7FFFu + ((u >> 16) & 1u);          // round-to-nearest-even
  return (unsigned short)(u >> 16);
}

// K-pair base for 16-bit A-matrix 16x32 fragment (contiguous order):
// lanes 0-15: V0..7 -> K=0..15 ; lanes 16-31: K=16..31.
__device__ __forceinline__ int a_kpair(int i, int g) {
  return 2 * i + 16 * g;
}

__device__ __forceinline__ v8f wmma_bf16(const FragB& a, const FragB& b, v8f c) {
  return __builtin_amdgcn_wmma_f32_16x16x32_bf16(false, a.v, false, b.v,
                                                 (short)0, c, false, false);
}

// ---------------------------------------------------------------------------
// f32 -> bf16 bulk convert
__global__ void cvt_bf16_kernel(const float* __restrict__ in,
                                unsigned short* __restrict__ out, int n) {
  int i = blockIdx.x * blockDim.x + threadIdx.x;
  int stride = gridDim.x * blockDim.x;
  for (; i < n; i += stride) out[i] = f2bf(in[i]);
}

// W[K=1024][N] f32 -> Wt[N][1024] bf16 (K-major so B-frags load b32 pairs).
// N is a power of two: pass log2(N) to keep index math shift/mask only.
__global__ void cvt_wT_kernel(const float* __restrict__ W,
                              unsigned short* __restrict__ Wt, int nshift) {
  const int N = 1 << nshift;
  int idx = blockIdx.x * blockDim.x + threadIdx.x;
  int total = DIM << nshift;
  int stride = gridDim.x * blockDim.x;
  for (; idx < total; idx += stride) {
    int k = idx >> nshift, n = idx & (N - 1);
    Wt[n * DIM + k] = f2bf(W[idx]);
  }
}

// bias depends only on delta = i - j >= 0 (causal T5 bucketing)
__global__ void bias_kernel(const float* __restrict__ rel_emb,
                            float* __restrict__ bias_tab) {
  int n = blockIdx.x * blockDim.x + threadIdx.x;
  if (n >= SEQ) return;
  int bucket;
  if (n < 16) {
    bucket = n;
  } else {
    int vl = 16 + (int)(logf((float)n * (1.0f / 16.0f)) * (16.0f / logf(8.0f)));
    bucket = vl < 31 ? vl : 31;
  }
  bias_tab[n] = rel_emb[bucket] * 11.3137084989848f;   // sqrt(128)
}

// ---------------------------------------------------------------------------
// Projection GEMM: Y = swish(xb · Wt^T + bias); 16x64 tile per wave,
// double-buffered K loop (loads for kk+32 issued before WMMAs of kk).
// mode 0: N=128, writes q/k bf16 (affine gamma/beta).
// mode 1: N=1024, writes v transposed bf16: vT[b][n][j].
template <int WN>
__global__ __launch_bounds__(256) void proj_kernel(
    const unsigned short* __restrict__ xb,   // [ROWS][DIM] bf16
    const unsigned short* __restrict__ Wt,   // [N][DIM]   bf16
    const float* __restrict__ bias,          // [N]
    int mode,
    const float* __restrict__ gamma, const float* __restrict__ beta,
    unsigned short* __restrict__ qb, unsigned short* __restrict__ kb,
    unsigned short* __restrict__ vT) {
  constexpr int WM = 8 / WN;
  const int tid = threadIdx.x;
  const int wave = tid >> 5, lane = tid & 31;
  const int g = lane >> 4, m = lane & 15;
  const int wm = wave / WN, wn = wave % WN;          // WN is constexpr: shifts
  const int row0 = blockIdx.x * (16 * WM) + wm * 16;
  const int col0 = blockIdx.y * (64 * WN) + wn * 64;

  const unsigned int* xb32 = (const unsigned int*)xb;
  const unsigned int* wt32 = (const unsigned int*)Wt;

  v8f acc[4];
#pragma unroll
  for (int t = 0; t < 4; ++t) acc[t] = v8f_zero();

  FragB a_cur, b_cur[4];
#pragma unroll
  for (int i = 0; i < 8; ++i)
    a_cur.u[i] = xb32[((row0 + m) * DIM + a_kpair(i, g)) >> 1];
#pragma unroll
  for (int t = 0; t < 4; ++t)
#pragma unroll
    for (int i = 0; i < 8; ++i)
      b_cur[t].u[i] = wt32[((col0 + t * 16 + m) * DIM + 2 * i + 16 * g) >> 1];

  for (int kk = 0; kk < DIM; kk += 32) {
    FragB a_nxt, b_nxt[4];
    const int kn = kk + 32;
    if (kn < DIM) {                      // issue next-step loads first
#pragma unroll
      for (int i = 0; i < 8; ++i)
        a_nxt.u[i] = xb32[((row0 + m) * DIM + kn + a_kpair(i, g)) >> 1];
#pragma unroll
      for (int t = 0; t < 4; ++t)
#pragma unroll
        for (int i = 0; i < 8; ++i)
          b_nxt[t].u[i] =
              wt32[((col0 + t * 16 + m) * DIM + kn + 2 * i + 16 * g) >> 1];
    }
#pragma unroll
    for (int t = 0; t < 4; ++t) acc[t] = wmma_bf16(a_cur, b_cur[t], acc[t]);
    a_cur = a_nxt;
#pragma unroll
    for (int t = 0; t < 4; ++t) b_cur[t] = b_nxt[t];
  }

  if (mode == 0) {
#pragma unroll
    for (int t = 0; t < 4; ++t) {
      int col = col0 + t * 16 + m;
      float bq = bias[col];
      float g0 = gamma[col], b0 = beta[col];
      float g1 = gamma[DQK + col], b1 = beta[DQK + col];
#pragma unroll
      for (int r = 0; r < 8; ++r) {
        int row = row0 + r + 8 * g;
        float y = acc[t][r] + bq;
        float s = y / (1.0f + __expf(-y));          // swish
        qb[row * DQK + col] = f2bf(s * g0 + b0);
        kb[row * DQK + col] = f2bf(s * g1 + b1);
      }
    }
  } else {
#pragma unroll
    for (int t = 0; t < 4; ++t) {
      int col = col0 + t * 16 + m;
      float bq = bias[col];
#pragma unroll
      for (int r = 0; r < 8; ++r) {
        int row = row0 + r + 8 * g;
        float y = acc[t][r] + bq;
        float s = y / (1.0f + __expf(-y));
        int b_idx = row >> 11, j = row & (SEQ - 1);
        vT[(size_t)b_idx * DV * SEQ + (size_t)col * SEQ + j] = f2bf(s);
      }
    }
  }
}

// ---------------------------------------------------------------------------
// Flash attention: one WG (8 wave32) per (batch, 16-row i-tile).
// Per group of 8 j-tiles: wave w builds attn tile for j-tile (grp*8+w) into
// LDS (bf16, 16x128); barrier; each wave then multiplies the shared attn
// tile against its private 128-column slice of vT, accumulating f32.
// The 32-WMMA consume loop double-buffers the v B-fragment; consecutive
// steps hit different accumulators, spacing same-acc WMMAs 8 apart.
__global__ __launch_bounds__(256) void attn_kernel(
    const unsigned short* __restrict__ qb,   // [ROWS][DQK] bf16
    const unsigned short* __restrict__ kb,   // [ROWS][DQK] bf16
    const unsigned short* __restrict__ vT,   // [B][DV][SEQ] bf16
    const float* __restrict__ bias_tab,      // [SEQ]
    float* __restrict__ out) {               // [B][SEQ][DV] f32
  const int tid = threadIdx.x;
  const int wave = tid >> 5, lane = tid & 31;
  const int g = lane >> 4, m = lane & 15;
  const int bt = blockIdx.x >> 7;            // batch
  const int it = blockIdx.x & 127;           // i-tile
  const int i0 = it * 16;

  __shared__ unsigned int attn_lds_u[16 * 64];   // 16 x 128 bf16, 4 KB
  unsigned short* attn_lds = (unsigned short*)attn_lds_u;

  const unsigned int* q32 = (const unsigned int*)qb;
  const unsigned int* k32 = (const unsigned int*)kb;
  const unsigned int* v32 = (const unsigned int*)vT;

  // q fragments for this row tile live in registers across the whole j-loop
  FragB qf[4];
#pragma unroll
  for (int c = 0; c < 4; ++c)
#pragma unroll
    for (int i = 0; i < 8; ++i)
      qf[c].u[i] = q32[((bt * SEQ + i0 + m) * DQK + c * 32 + a_kpair(i, g)) >> 1];

  v8f acc[8];
#pragma unroll
  for (int t = 0; t < 8; ++t) acc[t] = v8f_zero();

  const int ngroups = (it >> 3) + 1;          // causal: j-tiles <= it only
  const float scale = 1.0f / (float)SEQ;

  for (int grp = 0; grp < ngroups; ++grp) {
    // ---- sim tile for this wave's j-tile ----
    const int j0 = (grp * 8 + wave) * 16;
    if (grp + 1 < ngroups)                    // global_prefetch_b8 next k rows
      __builtin_prefetch(&kb[(bt * SEQ + j0 + 128 + m) * DQK], 0, 1);

    // preload all 4 k fragments, then run the 4-WMMA chain
    FragB kf[4];
#pragma unroll
    for (int c = 0; c < 4; ++c)
#pragma unroll
      for (int i = 0; i < 8; ++i)
        kf[c].u[i] =
            k32[((bt * SEQ + j0 + m) * DQK + c * 32 + 2 * i + 16 * g) >> 1];
    v8f s = v8f_zero();
#pragma unroll
    for (int c = 0; c < 4; ++c) s = wmma_bf16(qf[c], kf[c], s);

    // scale + rel-pos bias + laplacian(erf) + causal mask -> LDS bf16
#pragma unroll
    for (int r = 0; r < 8; ++r) {
      int i_abs = i0 + r + 8 * g;
      int j_abs = j0 + m;
      int d = i_abs - j_abs;
      float a = 0.0f;
      if (d >= 0) {
        float sv = s[r] * scale + bias_tab[d];
        a = 0.5f * (1.0f + erff((sv - 0.70710678118f) * 0.79788456080f));
      }
      attn_lds[(r + 8 * g) * 128 + wave * 16 + m] = f2bf(a);
    }
    __syncthreads();

    // ---- attn(16x128) @ v(128 x 128-col slice per wave) ----
    const int jb = grp * 128;
    const size_t vrow0 = ((size_t)bt * DV + wave * 128 + m) * SEQ;

    FragB af[4];                              // all A frags from LDS up front
#pragma unroll
    for (int h = 0; h < 4; ++h)
#pragma unroll
      for (int i = 0; i < 8; ++i)
        af[h].u[i] = attn_lds_u[(m * 128 + h * 32 + a_kpair(i, g)) >> 1];

    FragB b_cur, b_nxt;
#pragma unroll
    for (int i = 0; i < 8; ++i)               // step 0: (h=0,t=0)
      b_cur.u[i] = v32[(vrow0 + jb + 2 * i + 16 * g) >> 1];

#pragma unroll
    for (int step = 0; step < 32; ++step) {   // flattened (h,t)
      const int h = step >> 3, t = step & 7;
      if (step + 1 < 32) {
        const int hn = (step + 1) >> 3, tn = (step + 1) & 7;
#pragma unroll
        for (int i = 0; i < 8; ++i)
          b_nxt.u[i] = v32[(vrow0 + (size_t)tn * 16 * SEQ + jb + hn * 32 +
                            2 * i + 16 * g) >> 1];
      }
      acc[t] = wmma_bf16(af[h], b_cur, acc[t]);
      b_cur = b_nxt;
    }
    __syncthreads();
  }

  // epilogue: f32 out, C/D layout M = r + 8g, N = lane%16
#pragma unroll
  for (int t = 0; t < 8; ++t) {
    int col = wave * 128 + t * 16 + m;
#pragma unroll
    for (int r = 0; r < 8; ++r) {
      int row = i0 + r + 8 * g;
      out[((size_t)bt * SEQ + row) * DV + col] = acc[t][r];
    }
  }
}

// ---------------------------------------------------------------------------
extern "C" void kernel_launch(void* const* d_in, const int* in_sizes, int n_in,
                              void* d_out, int out_size, void* d_ws, size_t ws_size,
                              hipStream_t stream) {
  (void)in_sizes; (void)n_in; (void)out_size; (void)ws_size;
  const float* x     = (const float*)d_in[0];
  const float* Wqk   = (const float*)d_in[1];
  const float* bqk   = (const float*)d_in[2];
  const float* gamma = (const float*)d_in[3];
  const float* beta  = (const float*)d_in[4];
  const float* Wv    = (const float*)d_in[5];
  const float* bv    = (const float*)d_in[6];
  const float* rel   = (const float*)d_in[7];
  float* out = (float*)d_out;

  // workspace layout (~74.3 MB total)
  char* ws = (char*)d_ws;
  size_t off = 0;
  unsigned short* xb   = (unsigned short*)(ws + off); off += (size_t)ROWS * DIM * 2;
  unsigned short* wqkT = (unsigned short*)(ws + off); off += (size_t)DQK * DIM * 2;
  unsigned short* wvT  = (unsigned short*)(ws + off); off += (size_t)DV * DIM * 2;
  unsigned short* qb   = (unsigned short*)(ws + off); off += (size_t)ROWS * DQK * 2;
  unsigned short* kb   = (unsigned short*)(ws + off); off += (size_t)ROWS * DQK * 2;
  unsigned short* vT   = (unsigned short*)(ws + off); off += (size_t)BATCH * DV * SEQ * 2;
  float* bias_tab      = (float*)(ws + off);          off += (size_t)SEQ * 4;

  cvt_bf16_kernel<<<2048, 256, 0, stream>>>(x, xb, ROWS * DIM);
  cvt_wT_kernel<<<512, 256, 0, stream>>>(Wqk, wqkT, 7);    // N=128
  cvt_wT_kernel<<<2048, 256, 0, stream>>>(Wv, wvT, 10);    // N=1024
  bias_kernel<<<8, 256, 0, stream>>>(rel, bias_tab);

  // qk projection: N=128; 8 waves as 4(M) x 2(N) -> 64x128 block tile
  dim3 gq(ROWS / 64, 1);
  proj_kernel<2><<<gq, 256, 0, stream>>>(xb, wqkT, bqk, /*mode=*/0,
                                         gamma, beta, qb, kb, nullptr);
  // v projection: N=1024; 8 waves as 2(M) x 4(N) -> 32x256 block tile
  dim3 gv(ROWS / 32, DV / 256);
  proj_kernel<4><<<gv, 256, 0, stream>>>(xb, wvT, bv, /*mode=*/1,
                                         nullptr, nullptr, nullptr, nullptr, vT);

  attn_kernel<<<BATCH * (SEQ / 16), 256, 0, stream>>>(qb, kb, vT, bias_tab, out);
}
